// MAAB_61881888801146
// MI455X (gfx1250) — matrix operand order repaired and checked by reading, not compiled, not run on hardware
//
#include <hip/hip_runtime.h>
#include <hip/hip_bf16.h>

typedef __attribute__((ext_vector_type(16))) __bf16 v16bf;
typedef __attribute__((ext_vector_type(8)))  float  v8f;

union Frag {
    v16bf v;
    uint4 q[2];
    unsigned short u[16];
};

__device__ __forceinline__ unsigned short f2bf(float f) {
    unsigned int u = __float_as_uint(f);
    u += 0x7FFFu + ((u >> 16) & 1u);        // round-to-nearest-even
    return (unsigned short)(u >> 16);
}
__device__ __forceinline__ float bf2f(unsigned short s) {
    return __uint_as_float(((unsigned int)s) << 16);
}

// ---------------------------------------------------------------- convert
__global__ void k_f32_to_bf16(const float* __restrict__ src,
                              unsigned short* __restrict__ dst, int n) {
    int i = blockIdx.x * blockDim.x + threadIdx.x;
    if (i < n) dst[i] = f2bf(src[i]);
}

// ---------------------------------------------------------------- direct 3x3 SAME conv via WMMA
// in  : bf16 [B, CIN, H, W]
// wgt : bf16 [COUT, CIN, 3, 3]
// out : bf16 [B, COUT, H, W]
// block = 256 threads (8 waves); workgroup tile = 16 out-ch x 128 pixels.
// All dims are template constants -> pure shift/constant addressing.
// LDS input slab padded one column left/right -> branch-free 9-tap WMMA loop.
// global_prefetch of the next channel slab overlaps HBM with the WMMA pipe.
template <int H, int W, int CIN, int COUT>
__global__ __launch_bounds__(256) void k_conv3x3_wmma(
    const unsigned short* __restrict__ in, const unsigned short* __restrict__ wgt,
    unsigned short* __restrict__ out) {
    extern __shared__ unsigned short smem[];
    constexpr int PW          = W + 2;          // padded width
    constexpr int ntiles      = (H * W) >> 7;   // 128 px per workgroup
    constexpr int mtiles      = COUT >> 4;
    constexpr int rowsPerTile = 128 / W;
    constexpr int slabRows    = rowsPerTile + 2;
    constexpr int S           = slabRows * PW * 32;   // input slab elems
    constexpr int WS          = 16 * 9 * 32;          // weight slab elems

    const int tid  = threadIdx.x;
    const int lane = tid & 31;
    const int wave = tid >> 5;
    int bid = blockIdx.x;
    const int nt = bid % ntiles;  bid /= ntiles;
    const int mt = bid % mtiles;
    const int b  = bid / mtiles;
    const int y0 = nt * rowsPerTile;

    unsigned short* inSlab = smem;               // [slabRows][PW][32c]
    unsigned short* wSlab  = smem + S;           // [16oc][9tap][32c]

    constexpr int rowTiles = W >> 4;
    const int waveRow  = wave / rowTiles;
    const int wx0      = (wave % rowTiles) << 4;
    const int l16      = lane & 15;
    const int h        = lane >> 4;

    v8f acc = {};
    for (int c0 = 0; c0 < CIN; c0 += 32) {
        // stage input slab (vertical halo + 1-px horizontal zero pad)
        for (int i = tid; i < S; i += 256) {
            int xp = i % PW;             // padded x: actual x = xp-1
            int rc = i / PW;
            int c  = rc & 31;
            int r  = rc >> 5;
            int y  = y0 - 1 + r;
            int x  = xp - 1;
            unsigned short v = 0;
            if (y >= 0 && y < H && x >= 0 && x < W)
                v = in[(((size_t)b * CIN + c0 + c) * H + y) * W + x];
            inSlab[(r * PW + xp) * 32 + c] = v;
        }
        // stage weights [mo][tap][c]
        for (int i = tid; i < WS; i += 256) {
            int c   = i & 31;
            int tap = (i >> 5) % 9;
            int mo  = i / 288;
            wSlab[i] = wgt[(((size_t)(mt * 16 + mo)) * CIN + c0 + c) * 9 + tap];
        }
        __syncthreads();

        // prefetch next K-step's input slab into cache while WMMAs run
        if (c0 + 32 < CIN) {
            int y = y0 - 1 + (tid >> 5);               // 8 rows covered by waves
            int c = lane;                              // 32 channels by lanes
            if (y >= 0 && y < H && (tid >> 5) < slabRows)
                __builtin_prefetch(
                    &in[(((size_t)b * CIN + c0 + 32 + c) * H + y) * W], 0, 1);
        }

        #pragma unroll
        for (int tap = 0; tap < 9; ++tap) {
            const int ky = tap / 3, kx = tap % 3;
            Frag A, Bf;
            // A: 16x32 weights, lane M = l16, K(=c) contiguous
            const unsigned short* pa = wSlab + (l16 * 9 + tap) * 32;
            A.q[0] = *(const uint4*)(pa + 16 * h);
            A.q[1] = *(const uint4*)(pa + 16 * h + 8);
            // B: 32x16 shifted pixels (padded -> unconditional), K(=c) contiguous
            const unsigned short* pb =
                inSlab + ((waveRow + ky) * PW + wx0 + l16 + kx) * 32 + 16 * h;
            Bf.q[0] = *(const uint4*)(pb);
            Bf.q[1] = *(const uint4*)(pb + 8);
            acc = __builtin_amdgcn_wmma_f32_16x16x32_bf16(
                false, A.v, false, Bf.v, (short)0, acc, false, false);
        }
        __syncthreads();
    }
    const int y = y0 + waveRow;
    #pragma unroll
    for (int r = 0; r < 8; ++r) {
        int M  = r + 8 * h;
        int oc = mt * 16 + M;
        out[(((size_t)b * COUT + oc) * H + y) * W + wx0 + l16] = f2bf(acc[r]);
    }
}

// ---------------------------------------------------------------- fused attention
// kbf [B,32,1024], qbf [B,32,4096], vbf [B,256,1024] -> out f32 [B,256,4096] = v @ softmax_n(k^T q)
// one workgroup per (b, 32 columns of N1); ~260KB LDS (CDNA5 WGP has 320KB)
__global__ __launch_bounds__(256) void k_attention_wmma(
    const unsigned short* __restrict__ kbf, const unsigned short* __restrict__ qbf,
    const unsigned short* __restrict__ vbf, float* __restrict__ out) {
    extern __shared__ unsigned char smraw[];
    unsigned short* kT = (unsigned short*)smraw;          // [1024 n][32 c]
    unsigned short* qT = kT + 1024 * 32;                  // [32 m][32 c]
    unsigned short* pT = qT + 32 * 32;                    // [32 m][1024 n] bf16
    float* scores = (float*)(pT + 32 * 1024);             // [1024 n][32 m]
    float* red    = scores + 1024 * 32;                   // [8][32]
    const int tid  = threadIdx.x;
    const int lane = tid & 31;
    const int wave = tid >> 5;
    const int h    = lane >> 4;
    const int l16  = lane & 15;
    const int b  = blockIdx.x >> 7;
    const int m0 = (blockIdx.x & 127) << 5;

    for (int i = tid; i < 32 * 1024; i += 256) {
        int n = i & 1023, c = i >> 10;
        kT[n * 32 + c] = kbf[((size_t)b * 32 + c) * 1024 + n];
    }
    for (int i = tid; i < 32 * 32; i += 256) {
        int m = i & 31, c = i >> 5;
        qT[m * 32 + c] = qbf[((size_t)b * 32 + c) * 4096 + m0 + m];
    }
    // warm L2/WGP$ for the v rows this block will stream in phase 3
    __builtin_prefetch(&vbf[((size_t)b * 256 + tid) * 1024], 0, 1);
    __syncthreads();

    // phase 1: scores[n][m] = sum_c k[c,n] q[c,m]   (K = 32 -> one WMMA per tile)
    for (int t = wave; t < 128; t += 8) {
        int n16 = (t & 63) << 4;
        int m16 = (t >> 6) << 4;
        Frag A, Bf;
        const unsigned short* pa = kT + (n16 + l16) * 32;
        A.q[0] = *(const uint4*)(pa + 16 * h);
        A.q[1] = *(const uint4*)(pa + 16 * h + 8);
        const unsigned short* pb = qT + (m16 + l16) * 32 + 16 * h;
        Bf.q[0] = *(const uint4*)(pb);
        Bf.q[1] = *(const uint4*)(pb + 8);
        v8f acc = {};
        acc = __builtin_amdgcn_wmma_f32_16x16x32_bf16(
            false, A.v, false, Bf.v, (short)0, acc, false, false);
        #pragma unroll
        for (int r = 0; r < 8; ++r)
            scores[(n16 + r + 8 * h) * 32 + m16 + l16] = acc[r];
    }
    __syncthreads();

    // phase 2: softmax over n (1024) per column m; emit transposed bf16 p
    {
        const int m = tid & 31;
        const int part = tid >> 5;
        float lmax = -1e30f;
        for (int n = part; n < 1024; n += 8)
            lmax = fmaxf(lmax, scores[n * 32 + m]);
        red[part * 32 + m] = lmax;
        __syncthreads();
        if (tid < 32) {
            float mx = red[tid];
            for (int p = 1; p < 8; ++p) mx = fmaxf(mx, red[p * 32 + tid]);
            red[tid] = mx;
        }
        __syncthreads();
        const float mx = red[m];
        __syncthreads();
        float lsum = 0.f;
        for (int n = part; n < 1024; n += 8) {
            float e = __expf(scores[n * 32 + m] - mx);
            scores[n * 32 + m] = e;
            lsum += e;
        }
        red[part * 32 + m] = lsum;
        __syncthreads();
        if (tid < 32) {
            float s = 0.f;
            for (int p = 0; p < 8; ++p) s += red[p * 32 + tid];
            red[tid] = 1.f / s;
        }
        __syncthreads();
        const float inv = red[m];
        for (int n = part; n < 1024; n += 8)
            pT[m * 1024 + n] = f2bf(scores[n * 32 + m] * inv);
    }
    __syncthreads();

    // phase 3: att[c,m] = sum_n v[c,n] p[n,m]; A streamed from global, B from LDS
    for (int t = wave; t < 32; t += 8) {
        int c16 = (t & 15) << 4;
        int m16 = (t >> 4) << 4;
        v8f acc = {};
        const unsigned short* vbase = vbf + ((size_t)b * 256 + c16 + l16) * 1024;
        for (int k0 = 0; k0 < 1024; k0 += 32) {
            Frag A, Bf;
            A.q[0] = *(const uint4*)(vbase + k0 + 16 * h);
            A.q[1] = *(const uint4*)(vbase + k0 + 16 * h + 8);
            const unsigned short* pb = pT + (m16 + l16) * 1024 + k0 + 16 * h;
            Bf.q[0] = *(const uint4*)(pb);
            Bf.q[1] = *(const uint4*)(pb + 8);
            if (k0 + 128 < 1024)
                __builtin_prefetch(vbase + k0 + 128, 0, 1);
            acc = __builtin_amdgcn_wmma_f32_16x16x32_bf16(
                false, A.v, false, Bf.v, (short)0, acc, false, false);
        }
        #pragma unroll
        for (int r = 0; r < 8; ++r)
            out[((size_t)b * 256 + c16 + r + 8 * h) * 4096 + m0 + m16 + l16] = acc[r];
    }
}

// ---------------------------------------------------------------- channel branch (f32 glue)
__global__ void k_cf_logits(const float* __restrict__ x1, const float* __restrict__ wc,
                            float* __restrict__ cf, int total) {
    int idx = blockIdx.x * blockDim.x + threadIdx.x;
    if (idx >= total) return;
    int b = idx >> 12, n = idx & 4095;
    const float* xp = x1 + (size_t)b * 256 * 4096 + n;
    float s = 0.f;
    for (int c = 0; c < 256; ++c) s += wc[c] * xp[(size_t)c * 4096];
    cf[idx] = s;
}

__global__ __launch_bounds__(256) void k_softmax_n(float* __restrict__ cf, int N) {
    __shared__ float red[256];
    float* p = cf + (size_t)blockIdx.x * N;
    int tid = threadIdx.x;
    float lmax = -1e30f;
    for (int n = tid; n < N; n += 256) lmax = fmaxf(lmax, p[n]);
    red[tid] = lmax; __syncthreads();
    for (int s = 128; s > 0; s >>= 1) {
        if (tid < s) red[tid] = fmaxf(red[tid], red[tid + s]);
        __syncthreads();
    }
    float mx = red[0]; __syncthreads();
    float lsum = 0.f;
    for (int n = tid; n < N; n += 256) { float e = __expf(p[n] - mx); p[n] = e; lsum += e; }
    red[tid] = lsum; __syncthreads();
    for (int s = 128; s > 0; s >>= 1) {
        if (tid < s) red[tid] += red[tid + s];
        __syncthreads();
    }
    float inv = 1.f / red[0];
    for (int n = tid; n < N; n += 256) p[n] *= inv;
}

__global__ __launch_bounds__(256) void k_pool(const unsigned short* __restrict__ chf,
                                              const float* __restrict__ cf,
                                              float* __restrict__ pool) {
    __shared__ float red[256];
    const unsigned short* xp = chf + (size_t)blockIdx.x * 4096;
    const float* cp = cf + (size_t)(blockIdx.x >> 8) * 4096;
    int tid = threadIdx.x;
    float s = 0.f;
    for (int n = tid; n < 4096; n += 256) s += bf2f(xp[n]) * cp[n];
    red[tid] = s; __syncthreads();
    for (int st = 128; st > 0; st >>= 1) {
        if (tid < st) red[tid] += red[tid + st];
        __syncthreads();
    }
    if (tid == 0) pool[blockIdx.x] = red[0];
}

__global__ __launch_bounds__(256) void k_mlp(const float* __restrict__ pool,
    const float* __restrict__ wt1, const float* __restrict__ bt1,
    const float* __restrict__ ln_g, const float* __restrict__ ln_b,
    const float* __restrict__ wt2, const float* __restrict__ bt2,
    float* __restrict__ cw) {
    __shared__ float sp[256];
    __shared__ float st[32];
    __shared__ float stats[2];
    int b = blockIdx.x, tid = threadIdx.x;
    sp[tid] = pool[b * 256 + tid];
    __syncthreads();
    if (tid < 32) {
        float s = bt1[tid];
        for (int c = 0; c < 256; ++c) s += wt1[tid * 256 + c] * sp[c];
        st[tid] = s;
    }
    __syncthreads();
    if (tid == 0) {
        float mu = 0.f;
        for (int j = 0; j < 32; ++j) mu += st[j];
        mu *= (1.f / 32.f);
        float var = 0.f;
        for (int j = 0; j < 32; ++j) { float d = st[j] - mu; var += d * d; }
        var *= (1.f / 32.f);
        stats[0] = mu; stats[1] = rsqrtf(var + 1e-5f);
    }
    __syncthreads();
    if (tid < 32) {
        float t = (st[tid] - stats[0]) * stats[1] * ln_g[tid] + ln_b[tid];
        st[tid] = fmaxf(t, 0.f);
    }
    __syncthreads();
    float s = bt2[tid];
    for (int j = 0; j < 32; ++j) s += wt2[tid * 32 + j] * st[j];
    cw[b * 256 + tid] = s;
}

__global__ void k_finalize(float* __restrict__ out, const float* __restrict__ x1,
                           const float* __restrict__ cw, int total) {
    int idx = blockIdx.x * blockDim.x + threadIdx.x;
    if (idx >= total) return;
    int bc = idx >> 12;            // b*256 + c
    out[idx] += x1[idx] * (1.f + cw[bc]);
}

// ---------------------------------------------------------------- launch
extern "C" void kernel_launch(void* const* d_in, const int* in_sizes, int n_in,
                              void* d_out, int out_size, void* d_ws, size_t ws_size,
                              hipStream_t stream) {
    (void)in_sizes; (void)n_in; (void)out_size; (void)ws_size;
    const float* x1   = (const float*)d_in[0];
    const float* x2   = (const float*)d_in[1];
    const float* wq   = (const float*)d_in[2];
    const float* wk   = (const float*)d_in[3];
    const float* wv   = (const float*)d_in[4];
    const float* wc   = (const float*)d_in[5];
    const float* wch  = (const float*)d_in[6];
    const float* wt1  = (const float*)d_in[7];
    const float* bt1  = (const float*)d_in[8];
    const float* ln_g = (const float*)d_in[9];
    const float* ln_b = (const float*)d_in[10];
    const float* wt2  = (const float*)d_in[11];
    const float* bt2  = (const float*)d_in[12];
    float* out = (float*)d_out;

    const int B = 8, C1 = 256, C2 = 512;
    const int N1 = 4096, N2 = 1024;

    char* ws = (char*)d_ws;
    size_t off = 0;
    auto alloc = [&](size_t bytes) {
        char* p = ws + off;
        off = (off + bytes + 255) & ~(size_t)255;
        return p;
    };
    unsigned short* x1bf  = (unsigned short*)alloc((size_t)B * C1 * N1 * 2);
    unsigned short* x2bf  = (unsigned short*)alloc((size_t)B * C2 * N2 * 2);
    unsigned short* wqbf  = (unsigned short*)alloc((size_t)32 * C1 * 9 * 2);
    unsigned short* wkbf  = (unsigned short*)alloc((size_t)32 * C2 * 9 * 2);
    unsigned short* wvbf  = (unsigned short*)alloc((size_t)C1 * C2 * 9 * 2);
    unsigned short* wchbf = (unsigned short*)alloc((size_t)C1 * C1 * 9 * 2);
    unsigned short* qbf   = (unsigned short*)alloc((size_t)B * 32 * N1 * 2);
    unsigned short* kbf   = (unsigned short*)alloc((size_t)B * 32 * N2 * 2);
    unsigned short* vbf   = (unsigned short*)alloc((size_t)B * C1 * N2 * 2);
    unsigned short* chfbf = (unsigned short*)alloc((size_t)B * C1 * N1 * 2);
    float* cf   = (float*)alloc((size_t)B * N1 * 4);
    float* pool = (float*)alloc((size_t)B * C1 * 4);
    float* cw   = (float*)alloc((size_t)B * C1 * 4);

    auto cvt = [&](const float* s, unsigned short* d, int n) {
        k_f32_to_bf16<<<(n + 255) / 256, 256, 0, stream>>>(s, d, n);
    };
    cvt(x1,  x1bf,  B * C1 * N1);
    cvt(x2,  x2bf,  B * C2 * N2);
    cvt(wq,  wqbf,  32 * C1 * 9);
    cvt(wk,  wkbf,  32 * C2 * 9);
    cvt(wv,  wvbf,  C1 * C2 * 9);
    cvt(wch, wchbf, C1 * C1 * 9);

    // LDS: padded input slab (rows+2)*(W+2)*32 + weight slab 16*9*32, bf16
    const int smem64 = (4 * 66 * 32 + 16 * 9 * 32) * 2;   // 26112 B (64x64)
    const int smem32 = (6 * 34 * 32 + 16 * 9 * 32) * 2;   // 22272 B (32x32)

    // q = conv3x3(x1, wq) : 256 -> 32, 64x64
    k_conv3x3_wmma<64, 64, 256, 32><<<B * 2 * (N1 / 128), 256, smem64, stream>>>(
        x1bf, wqbf, qbf);
    // k = conv3x3(x2, wk) : 512 -> 32, 32x32
    k_conv3x3_wmma<32, 32, 512, 32><<<B * 2 * (N2 / 128), 256, smem32, stream>>>(
        x2bf, wkbf, kbf);
    // v = conv3x3(x2, wv) : 512 -> 256, 32x32
    k_conv3x3_wmma<32, 32, 512, 256><<<B * 16 * (N2 / 128), 256, smem32, stream>>>(
        x2bf, wvbf, vbf);
    // chf = conv3x3(x1, wch) : 256 -> 256, 64x64
    k_conv3x3_wmma<64, 64, 256, 256><<<B * 16 * (N1 / 128), 256, smem64, stream>>>(
        x1bf, wchbf, chfbf);

    // fused attention: out = v @ softmax_n(k^T q)
    const int attn_smem = 1024 * 32 * 2 + 32 * 32 * 2 + 32 * 1024 * 2
                        + 1024 * 32 * 4 + 256 * 4;        // 265216 B
    k_attention_wmma<<<B * (N1 / 32), 256, attn_smem, stream>>>(kbf, qbf, vbf, out);

    // channel branch
    k_cf_logits<<<(B * N1 + 255) / 256, 256, 0, stream>>>(x1, wc, cf, B * N1);
    k_softmax_n<<<B, 256, 0, stream>>>(cf, N1);
    k_pool<<<B * C1, 256, 0, stream>>>(chfbf, cf, pool);
    k_mlp<<<B, 256, 0, stream>>>(pool, wt1, bt1, ln_g, ln_b, wt2, bt2, cw);

    // out += x1 * (1 + channel_weight)   (folds the "+ x1" from att_con)
    k_finalize<<<(B * C1 * N1 + 255) / 256, 256, 0, stream>>>(out, x1, cw, B * C1 * N1);
}
